// CombineLoss_28252294873499
// MI455X (gfx1250) — compile-verified, matrix operand order checked by reading, not run on hardware
//
#include <hip/hip_runtime.h>
#include <hip/hip_bf16.h>

// MI455X reasoning: workload is HBM-bound (~224 MiB input, ~10us at 23.3 TB/s).
// One fused streaming pass. target staged to LDS via gfx1250 async-to-LDS copy
// (reused ~30x by the 7x7 disk conv + gather). Disk conv = separable 7x7 box
// (two exact f16 Toeplitz WMMAs, v_wmma_f32_16x16x32_f16) minus 20 corner taps.
// Deterministic two-stage reduction (no float atomics).

typedef _Float16 h8  __attribute__((ext_vector_type(8)));
typedef _Float16 h16 __attribute__((ext_vector_type(16)));
typedef float    f8  __attribute__((ext_vector_type(8)));

#define TW 128          // tile width (pixels)
#define TH 32           // tile height (pixels)
#define RT 4            // row-tiles per block
#define RAW_COLS 144    // padded halo tile width (128+6 -> 144 for alignment)
#define RAW_ROWS 38     // TH + 2*3 halo
#define H7S 48          // padded per-column H7 stride (38 rows + zero pad)
#define EPSF 1e-10f

// LDS layout (phases separated by barriers):
//   [0      .. 21888) rawT   int32  38x144   (aliased by reduction buffer later)
//   [21888  .. 32832) ldsT   f16    38x144
//   [32832  .. 45120) h7t    f16    128x48   (transposed horizontal box sums)
//   [45120  .. 61504) atb    f32    32x128   (|t - ave|)
#define SMEM_BYTES 61504

__global__ __launch_bounds__(256) void
combine_loss_main(const float* __restrict__ pred,   // (2,8,2,1024,1024)
                  const float* __restrict__ diss,   // (1,8,2,1024,1024)
                  const int*   __restrict__ target, // (8,1024,1024)
                  float*       __restrict__ ws)     // 512 blocks * 12 partials
{
    __shared__ __align__(16) unsigned char smem[SMEM_BYTES];
    int*      rawT = (int*)smem;
    _Float16* ldsT = (_Float16*)(smem + 21888);
    _Float16* h7t  = (_Float16*)(smem + 32832);
    float*    atb  = (float*)(smem + 45120);
    float*    red  = (float*)smem;              // phase-4 alias over rawT

    const int b    = blockIdx.z;
    const int x0   = blockIdx.x * TW;
    const int tid  = threadIdx.x;
    const int lane = tid & 31;
    const int wave = tid >> 5;
    const int hiH  = (lane >> 4) & 1;           // lane >= 16
    const int ln   = lane & 15;                 // M (A) or N (B/D) index
    const int kb0  = hiH ? 8 : 0;               // f16 K-chunk bases per ISA layout
    const int kb1  = hiH ? 24 : 16;
    const int xs   = wave * 16;                 // this wave's 16-col subtile

    // Toeplitz operand: value(k,n)=1 iff n<=k<=n+6. Same per-lane packing is
    // simultaneously B of the horizontal pass and A of the vertical pass.
    h16 bT;
#pragma unroll
    for (int e = 0; e < 16; ++e) {
        int k = (e < 8) ? (kb0 + e) : (kb1 + e - 8);
        bT[e] = (k >= ln && k <= ln + 6) ? (_Float16)1.0f : (_Float16)0.0f;
    }

    float acc[12];
#pragma unroll
    for (int q = 0; q < 12; ++q) acc[q] = 0.0f;

    const unsigned long long tbase =
        (unsigned long long)(const void*)(target + (size_t)b * (1024 * 1024));

    for (int it = 0; it < RT; ++it) {
        const int y0 = (blockIdx.y * RT + it) * TH;

        // ---- phase 0: async-stage target tile (with zeroed SAME halo) ----
        for (int i = tid; i < RAW_ROWS * RAW_COLS; i += 256) {
            int r = i / RAW_COLS, c = i % RAW_COLS;
            int gy = y0 - 3 + r, gx = x0 - 3 + c;
            bool in = (c < TW + 6) & (gy >= 0) & (gy < 1024) & (gx >= 0) & (gx < 1024);
            if (in) {
                unsigned loff = (unsigned)(size_t)(void*)(&rawT[i]);
                unsigned goff = (unsigned)(gy * 1024 + gx) * 4u;
                asm volatile("global_load_async_to_lds_b32 %0, %1, %2"
                             :: "v"(loff), "v"(goff), "s"(tbase) : "memory");
            } else {
                rawT[i] = 0;
            }
        }
        asm volatile("s_wait_asynccnt 0" ::: "memory");
        __syncthreads();

        // ---- phase 1: int32 -> f16, zero H7 buffer (keeps x0 pads finite) ----
        for (int i = tid; i < RAW_ROWS * RAW_COLS; i += 256)
            ldsT[i] = (_Float16)(float)rawT[i];
        for (int i = tid; i < (128 * H7S) / 2; i += 256)
            ((unsigned*)h7t)[i] = 0u;
        __syncthreads();

        // ---- phase 2: disk conv via WMMA (per wave, self-contained subtile) ----
        const f8 zero8 = {0.f,0.f,0.f,0.f,0.f,0.f,0.f,0.f};
#pragma unroll
        for (int sy = 0; sy < 2; ++sy) {
            const int ys = sy * 16;

            // horizontal 7-box, rows ys-3..ys+12  (ldsT row = ys + M)
            h16 a;
            {
                const _Float16* base = &ldsT[(ys + ln) * RAW_COLS + xs];
                h8 lo = *(const h8*)(base + kb0);
                h8 hi = *(const h8*)(base + kb1);
#pragma unroll
                for (int e = 0; e < 8; ++e) { a[e] = lo[e]; a[e + 8] = hi[e]; }
            }
            f8 d1 = __builtin_amdgcn_wmma_f32_16x16x32_f16(
                        false, a, false, bT, (short)0, zero8, false, false);
#pragma unroll
            for (int e = 0; e < 8; ++e) {
                int M = e + (hiH ? 8 : 0);
                h7t[(xs + ln) * H7S + ys + M] = (_Float16)d1[e];
            }

            // horizontal 7-box, rows ys+3..ys+18 (store only the new rows 13..18)
            {
                const _Float16* base = &ldsT[(ys + 6 + ln) * RAW_COLS + xs];
                h8 lo = *(const h8*)(base + kb0);
                h8 hi = *(const h8*)(base + kb1);
#pragma unroll
                for (int e = 0; e < 8; ++e) { a[e] = lo[e]; a[e + 8] = hi[e]; }
            }
            f8 d2 = __builtin_amdgcn_wmma_f32_16x16x32_f16(
                        false, a, false, bT, (short)0, zero8, false, false);
#pragma unroll
            for (int e = 0; e < 8; ++e) {
                int M = e + (hiH ? 8 : 0);
                if (M >= 10) h7t[(xs + ln) * H7S + ys + 6 + M] = (_Float16)d2[e];
            }

            // vertical 7-box: A = Toeplitz (same packed values), B = H7 column
            h16 bv;
            {
                const _Float16* hb = &h7t[(xs + ln) * H7S + ys];
                h8 lo = *(const h8*)(hb + kb0);
                h8 hi = *(const h8*)(hb + kb1);
#pragma unroll
                for (int e = 0; e < 8; ++e) { bv[e] = lo[e]; bv[e + 8] = hi[e]; }
            }
            f8 V = __builtin_amdgcn_wmma_f32_16x16x32_f16(
                       false, bT, false, bv, (short)0, zero8, false, false);

            // corner corrections: disk = box49 - (H7(y∓3)-t(y∓3)) - 8 corner taps
#pragma unroll
            for (int e = 0; e < 8; ++e) {
                int M  = e + (hiH ? 8 : 0);
                int yy = ys + M;
                int xx = xs + ln;
                const _Float16* tb = &ldsT[(yy + 3) * RAW_COLS + (xx + 3)];
                float h7m = (float)h7t[xx * H7S + yy];
                float h7p = (float)h7t[xx * H7S + yy + 6];
                float tm3 = (float)tb[-3 * RAW_COLS];
                float tp3 = (float)tb[ 3 * RAW_COLS];
                float cr  = (float)tb[-2 * RAW_COLS - 3] + (float)tb[-2 * RAW_COLS + 3]
                          + (float)tb[ 2 * RAW_COLS - 3] + (float)tb[ 2 * RAW_COLS + 3]
                          + (float)tb[-1 * RAW_COLS - 3] + (float)tb[-1 * RAW_COLS + 3]
                          + (float)tb[ 1 * RAW_COLS - 3] + (float)tb[ 1 * RAW_COLS + 3];
                float sum29 = V[e] - (h7m - tm3) - (h7p - tp3) - cr;
                float ave   = sum29 * (1.0f / 29.0f);
                atb[yy * TW + xx] = fabsf((float)tb[0] - ave);
            }
        }
        __syncthreads();

        // ---- phase 3: fused focal/dice/edge accumulation (coalesced streams) ----
#pragma unroll
        for (int p = 0; p < (TH * TW) / 256; ++p) {
            int idx = p * 256 + tid;
            int yy = idx / TW, xx = idx % TW;
            size_t pix = (size_t)(y0 + yy) * 1024 + (size_t)(x0 + xx);
            int   ti = rawT[(yy + 3) * RAW_COLS + (xx + 3)];
            float tf = (float)ti;
            float at = atb[idx];
#pragma unroll
            for (int n = 0; n < 2; ++n) {
                const float* pb = pred + ((((size_t)(n * 8 + b)) * 2) << 20) + pix;
                float z0 = pb[0];
                float z1 = pb[1 << 20];
                float mx = fmaxf(z0, z1);
                float e0 = __expf(z0 - mx), e1 = __expf(z1 - mx);
                float inv = 1.0f / (e0 + e1);
                float p0 = e0 * inv, p1 = e1 * inv;
                float pt = ti ? p1 : p0;
                float lp = __logf(pt + EPSF);
                acc[0 + n] -= lp;
                acc[3 + n] -= lp * at;
                acc[6 + n] += p1 * tf;
                acc[9 + n] += p1 + tf;
            }
            {
                const float* db = diss + (((size_t)(b * 2)) << 20) + pix;
                float d0 = db[0], d1v = db[1 << 20];
                float pt = ti ? d1v : d0;
                float lp = __logf(pt + EPSF);
                acc[2]  -= lp;
                acc[5]  -= lp * at;
                acc[8]  += d1v * tf;
                acc[11] += d1v + tf;
            }
        }
        __syncthreads();   // protect LDS before next iteration / phase 4 alias
    }

    // ---- phase 4: deterministic in-block tree reduction -> workspace ----
#pragma unroll
    for (int q = 0; q < 12; ++q) red[tid * 12 + q] = acc[q];
    __syncthreads();
    for (int s = 128; s > 0; s >>= 1) {
        if (tid < s) {
#pragma unroll
            for (int q = 0; q < 12; ++q) red[tid * 12 + q] += red[(tid + s) * 12 + q];
        }
        __syncthreads();
    }
    if (tid == 0) {
        int blockFlat = (blockIdx.z * 8 + blockIdx.y) * 8 + blockIdx.x; // = b*64 + k
        float* w = ws + (size_t)blockFlat * 12;
#pragma unroll
        for (int q = 0; q < 12; ++q) w[q] = red[q];
    }
}

__global__ __launch_bounds__(128) void
combine_loss_finalize(const float* __restrict__ ws,
                      const float* __restrict__ sigma,
                      const float* __restrict__ diff,
                      float*       __restrict__ out)
{
    __shared__ float sr[96];
    int t = threadIdx.x;
    if (t < 96) {
        int q = t / 8, b = t % 8;
        float s = 0.0f;
        for (int k = 0; k < 64; ++k)           // fixed order -> deterministic
            s += ws[(size_t)(b * 64 + k) * 12 + q];
        sr[t] = s;
    }
    __syncthreads();
    if (t == 0) {
        const float inv_px = 1.0f / (8.0f * 1024.0f * 1024.0f);
        float s0 = sigma[0] * sigma[0];
        float s1 = sigma[1] * sigma[1];
        float s2 = sigma[2] * sigma[2];
        float loss = 0.0f;
        for (int n = 0; n < 3; ++n) {
            float F = 0.f, E = 0.f, dsum = 0.f;
            for (int b = 0; b < 8; ++b) {
                F += sr[(0 + n) * 8 + b];
                E += sr[(3 + n) * 8 + b];
                float I = sr[(6 + n) * 8 + b];
                float U = sr[(9 + n) * 8 + b];
                dsum += 2.0f * I / (U + EPSF);
            }
            float focal = F * inv_px;
            float edge  = E * inv_px;
            float dice  = 1.0f - dsum * 0.125f;
            loss += focal / s0 + dice / s1 + edge / s2;
        }
        loss += diff[0] + 0.5f * (__logf(s0) + __logf(s1) + __logf(s2));
        out[0] = loss;
    }
}

extern "C" void kernel_launch(void* const* d_in, const int* in_sizes, int n_in,
                              void* d_out, int out_size, void* d_ws, size_t ws_size,
                              hipStream_t stream) {
    const float* pred   = (const float*)d_in[0];   // (2,8,2,1024,1024) f32
    const float* diss   = (const float*)d_in[1];   // (1,8,2,1024,1024) f32
    const int*   target = (const int*)d_in[2];     // (8,1024,1024) i32
    const float* diff   = (const float*)d_in[3];   // scalar
    const float* sigma  = (const float*)d_in[4];   // (3,)
    float* out = (float*)d_out;
    float* ws  = (float*)d_ws;                     // needs 512*12*4 = 24576 B

    dim3 grid(8, 8, 8);   // x-tiles, y-supertiles (4 row-tiles each), batch
    combine_loss_main<<<grid, 256, 0, stream>>>(pred, diss, target, ws);
    combine_loss_finalize<<<1, 128, 0, stream>>>(ws, sigma, diff, out);
}
